// FP8Linear_90305982366162
// MI455X (gfx1250) — 8-bit matrix operand order repaired and checked by reading, compiled, not run on hardware
//
#include <hip/hip_runtime.h>
#include <stdint.h>

typedef __attribute__((ext_vector_type(16))) int   v16i;
typedef __attribute__((ext_vector_type(8)))  float v8f;

struct __align__(16) U4 { unsigned x, y, z, w; };

__device__ __forceinline__ unsigned f32_to_e4m3(float f) {
    unsigned bits = __float_as_uint(f);
    unsigned sign = (bits >> 24) & 0x80u;
    unsigned ab   = bits & 0x7fffffffu;
    float a = __uint_as_float(ab);
    if (a != a)      return sign | 0x7Fu;
    if (a >= 448.0f) return sign | 0x7Eu;
    if (a < 0.015625f) {
        unsigned q = (unsigned)__float2int_rn(a * 512.0f);
        return sign | q;
    }
    unsigned lsb = (ab >> 20) & 1u;
    ab += 0x7FFFFu + lsb;
    unsigned e = ab >> 23;
    unsigned m = (ab >> 20) & 7u;
    int ee = (int)e - 127 + 7;
    if (ee >= 16) return sign | 0x7Eu;
    return sign | ((unsigned)ee << 3) | m;
}

__global__ void zero_amax(unsigned* a) {
    a[0] = 0u; a[1] = 0u;
}

__global__ void __launch_bounds__(256)
amax_kernel(const float* __restrict__ p, int n, unsigned* __restrict__ out) {
    float m = 0.0f;
    for (int i = blockIdx.x * blockDim.x + threadIdx.x; i < n;
         i += gridDim.x * blockDim.x)
        m = fmaxf(m, fabsf(p[i]));
    #pragma unroll
    for (int off = 16; off; off >>= 1)
        m = fmaxf(m, __shfl_xor(m, off, 32));
    __shared__ float sm[8];
    int lane = threadIdx.x & 31, w = threadIdx.x >> 5;
    if (lane == 0) sm[w] = m;
    __syncthreads();
    if (threadIdx.x == 0) {
        float mm = sm[0];
        #pragma unroll
        for (int i = 1; i < 8; ++i) mm = fmaxf(mm, sm[i]);
        atomicMax(out, __float_as_uint(mm));
    }
}

__global__ void __launch_bounds__(256)
quant_rows(const float* __restrict__ in, const unsigned* __restrict__ amax,
           unsigned* __restrict__ out, int n4) {
    int id = blockIdx.x * blockDim.x + threadIdx.x;
    if (id >= n4) return;
    float s = 448.0f / fmaxf(__uint_as_float(*amax), 1e-12f);
    const float* p = in + (size_t)id * 4;
    unsigned b0 = f32_to_e4m3(p[0] * s);
    unsigned b1 = f32_to_e4m3(p[1] * s);
    unsigned b2 = f32_to_e4m3(p[2] * s);
    unsigned b3 = f32_to_e4m3(p[3] * s);
    out[id] = b0 | (b1 << 8) | (b2 << 16) | (b3 << 24);
}

__global__ void __launch_bounds__(256)
quant_wT(const float* __restrict__ w, const unsigned* __restrict__ amax,
         unsigned* __restrict__ wt, int OUT, int K) {
    int id = blockIdx.x * blockDim.x + threadIdx.x;
    int kq = K >> 2;
    if (id >= OUT * kq) return;
    int n = id / kq;
    int k = (id - n * kq) * 4;
    float s = 448.0f / fmaxf(__uint_as_float(*amax), 1e-12f);
    unsigned b0 = f32_to_e4m3(w[(size_t)(k + 0) * OUT + n] * s);
    unsigned b1 = f32_to_e4m3(w[(size_t)(k + 1) * OUT + n] * s);
    unsigned b2 = f32_to_e4m3(w[(size_t)(k + 2) * OUT + n] * s);
    unsigned b3 = f32_to_e4m3(w[(size_t)(k + 3) * OUT + n] * s);
    wt[id] = b0 | (b1 << 8) | (b2 << 16) | (b3 << 24);
}

__device__ __forceinline__ void async_b128(unsigned lds_addr, const void* gptr) {
    unsigned long long ga = (unsigned long long)(size_t)gptr;
    asm volatile("global_load_async_to_lds_b128 %0, %1, off"
                 :: "v"(lds_addr), "v"(ga) : "memory");
}
__device__ __forceinline__ void async_b64(unsigned lds_addr, const void* gptr) {
    unsigned long long ga = (unsigned long long)(size_t)gptr;
    asm volatile("global_load_async_to_lds_b64 %0, %1, off"
                 :: "v"(lds_addr), "v"(ga) : "memory");
}
__device__ __forceinline__ void wait_async0() {
    asm volatile("s_wait_asynccnt 0x0" ::: "memory");
}

__device__ __forceinline__ void
stage_async(const unsigned char* __restrict__ xq,
            const unsigned char* __restrict__ wq,
            int K, int blockRow0, int col0, int k0,
            unsigned ldsB, unsigned ldsA, int tid) {
    #pragma unroll
    for (int j = 0; j < 2; ++j) {
        int c = tid + j * 256;
        int t = c >> 7;
        int l = (c >> 2) & 31;
        int g = c & 3;
        int n = col0 + t * 16 + (l & 15);
        const void* src = wq + (size_t)n * K + k0 + (l >> 4) * 16 + g * 32;
        async_b128(ldsB + (unsigned)c * 16u, src);
    }
    #pragma unroll
    for (int j = 0; j < 8; ++j) {
        int c = tid + j * 256;
        int w = c >> 8;
        int l = (c >> 3) & 31;
        int p = c & 7;
        int row = blockRow0 + w * 16 + (l & 15);
        const void* src = xq + (size_t)row * K + k0
                        + (p >> 1) * 32 + (l >> 4) * 16 + (p & 1) * 8;
        async_b64(ldsA + (unsigned)c * 8u, src);
    }
}

#define BUF_BYTES 24576
#define B_OFF     0
#define A_OFF     8192

__global__ void __launch_bounds__(256)
fp8_gemm_wmma(const unsigned char* __restrict__ xq,
              const unsigned char* __restrict__ wq,
              const unsigned* __restrict__ amax,
              const float* __restrict__ bias,
              float* __restrict__ out,
              int K, int OUT) {
    __shared__ __align__(128) unsigned char smem[2 * BUF_BYTES];

    const int tid  = threadIdx.x;
    const int lane = tid & 31;
    const int wave = tid >> 5;
    const int r    = lane & 15;
    const int hi   = lane >> 4;
    const int blockRow0 = blockIdx.y * 128;
    const int col0      = blockIdx.x * 64;
    const unsigned lbase = (unsigned)(size_t)&smem[0];

    v8f acc0 = {}, acc1 = {}, acc2 = {}, acc3 = {};

    const int NS = K / 128;
    stage_async(xq, wq, K, blockRow0, col0, 0,
                lbase + B_OFF, lbase + A_OFF, tid);

    for (int step = 0; step < NS; ++step) {
        const int      cur  = step & 1;
        const unsigned boff = (unsigned)cur * BUF_BYTES;

        wait_async0();
        __syncthreads();

        if (step + 1 < NS) {
            const unsigned noff = (unsigned)(1 - cur) * BUF_BYTES;
            stage_async(xq, wq, K, blockRow0, col0, (step + 1) * 128,
                        lbase + noff + B_OFF, lbase + noff + A_OFF, tid);
        }

        const U4* Af = (const U4*)(smem + boff + A_OFF + (wave * 32 + lane) * 64);
        U4 a0 = Af[0], a1 = Af[1], a2 = Af[2], a3 = Af[3];
        v16i A;
        A[0]=(int)a0.x;  A[1]=(int)a0.y;  A[2]=(int)a0.z;  A[3]=(int)a0.w;
        A[4]=(int)a1.x;  A[5]=(int)a1.y;  A[6]=(int)a1.z;  A[7]=(int)a1.w;
        A[8]=(int)a2.x;  A[9]=(int)a2.y;  A[10]=(int)a2.z; A[11]=(int)a2.w;
        A[12]=(int)a3.x; A[13]=(int)a3.y; A[14]=(int)a3.z; A[15]=(int)a3.w;

        #pragma unroll
        for (int t = 0; t < 4; ++t) {
            const U4* Bf = (const U4*)(smem + boff + B_OFF + (t * 32 + lane) * 64);
            U4 b0 = Bf[0], b1 = Bf[1], b2 = Bf[2], b3 = Bf[3];
            v16i B;
            B[0]=(int)b0.x;  B[1]=(int)b0.y;  B[2]=(int)b0.z;  B[3]=(int)b0.w;
            B[4]=(int)b1.x;  B[5]=(int)b1.y;  B[6]=(int)b1.z;  B[7]=(int)b1.w;
            B[8]=(int)b2.x;  B[9]=(int)b2.y;  B[10]=(int)b2.z; B[11]=(int)b2.w;
            B[12]=(int)b3.x; B[13]=(int)b3.y; B[14]=(int)b3.z; B[15]=(int)b3.w;
            v8f acc = (t == 0) ? acc0 : (t == 1) ? acc1 : (t == 2) ? acc2 : acc3;
            acc = __builtin_amdgcn_wmma_f32_16x16x128_fp8_fp8(A, B, (short)0,
                                                              acc, false, false);
            if      (t == 0) acc0 = acc;
            else if (t == 1) acc1 = acc;
            else if (t == 2) acc2 = acc;
            else             acc3 = acc;
        }
    }

    float ax = fmaxf(__uint_as_float(amax[0]), 1e-12f);
    float aw = fmaxf(__uint_as_float(amax[1]), 1e-12f);
    float factor = (ax * (1.0f / 448.0f)) * (aw * (1.0f / 448.0f));

    const int row0 = blockRow0 + wave * 16;
    v8f accs[4] = {acc0, acc1, acc2, acc3};
    #pragma unroll
    for (int t = 0; t < 4; ++t) {
        int n = col0 + t * 16 + r;
        float b = bias[n];
        #pragma unroll
        for (int j = 0; j < 8; ++j) {
            int row = row0 + hi * 8 + j;
            out[(size_t)row * OUT + n] = accs[t][j] * factor + b;
        }
    }
}

extern "C" void kernel_launch(void* const* d_in, const int* in_sizes, int n_in,
                              void* d_out, int out_size, void* d_ws, size_t ws_size,
                              hipStream_t stream) {
    const float* x    = (const float*)d_in[0];
    const float* w    = (const float*)d_in[1];
    const float* bias = (const float*)d_in[2];
    float* out = (float*)d_out;

    const int OUT  = in_sizes[2];
    const int K    = in_sizes[1] / OUT;
    const int ROWS = in_sizes[0] / K;

    unsigned*      amax = (unsigned*)d_ws;
    unsigned char* xq   = (unsigned char*)d_ws + 256;
    unsigned char* wq   = xq + (size_t)ROWS * K;

    zero_amax<<<1, 1, 0, stream>>>(amax);
    amax_kernel<<<1024, 256, 0, stream>>>(x, ROWS * K, amax + 0);
    amax_kernel<<<1024, 256, 0, stream>>>(w, K * OUT, amax + 1);

    int n4x = (ROWS * K) / 4;
    quant_rows<<<(n4x + 255) / 256, 256, 0, stream>>>(x, amax + 0,
                                                      (unsigned*)xq, n4x);
    int n4w = (OUT * K) / 4;
    quant_wT<<<(n4w + 255) / 256, 256, 0, stream>>>(w, amax + 1,
                                                    (unsigned*)wq, OUT, K);

    dim3 grid(OUT / 64, ROWS / 128);
    fp8_gemm_wmma<<<grid, 256, 0, stream>>>(xq, wq, amax, bias, out, K, OUT);
}
